// VanillaAttention_75754633167487
// MI455X (gfx1250) — matrix operand order repaired and checked by reading, not compiled, not run on hardware
//
#include <hip/hip_runtime.h>
#include <math.h>

// ---------------- problem constants ----------------
constexpr int BB   = 2;
constexpr int TT   = 4096;
constexpr int DD   = 768;
constexpr int HH   = 12;
constexpr int HD   = 64;
constexpr int MTOT = BB * TT;     // 8192 rows of x
constexpr int NQKV = 3 * DD;      // 2304
constexpr float SCALE = 0.125f;   // 1/sqrt(64)

typedef __attribute__((ext_vector_type(16))) __bf16 v16bf;
typedef __attribute__((ext_vector_type(8)))  float  v8f;

union V16U { uint4 q[2]; v16bf v; };

__device__ inline v8f v8f_zero() {
  v8f z; for (int i = 0; i < 8; ++i) z[i] = 0.0f; return z;
}

// fp32 -> bf16 round-to-nearest-even
__device__ inline unsigned short f2bf_bits(float f) {
  union { float f; unsigned u; } v; v.f = f;
  unsigned r = v.u + 0x7FFFu + ((v.u >> 16) & 1u);
  return (unsigned short)(r >> 16);
}
__device__ inline __bf16 f2bf(float f) {
  union { unsigned short s; __bf16 b; } u; u.s = f2bf_bits(f); return u.b;
}

// ---- WMMA fragment loaders (ISA 7.12.2 layouts, wave32) ----
// A (16x32 bf16, row-major source, leading dim ld):
//   lane L: row M = L%16 ; K = (L/16)*8 + {0..7} in v0..3, +16 in v4..7
__device__ inline v16bf load_a_frag(const __bf16* p, int ld) {
  int lane = threadIdx.x & 31;
  const __bf16* q = p + (lane & 15) * ld + ((lane >> 4) << 3);
  V16U u;
  u.q[0] = *(const uint4*)(q);        // K = klo .. klo+7
  u.q[1] = *(const uint4*)(q + 16);   // K = klo+16 .. klo+23
  return u.v;
}
// B (32x16 bf16) read from an N-major (transposed) source, leading dim ld:
//   lane L: column N = L%16 ; K = (L/16)*8 + {0..7} in v0..3, +16 in v4..7
__device__ inline v16bf load_bt_frag(const __bf16* p, int ld) {
  int lane = threadIdx.x & 31;
  const __bf16* q = p + (lane & 15) * ld + ((lane >> 4) << 3);
  V16U u;
  u.q[0] = *(const uint4*)(q);
  u.q[1] = *(const uint4*)(q + 16);
  return u.v;
}

__device__ inline v8f wmma_bf16(v16bf a, v16bf b, v8f c) {
  return __builtin_amdgcn_wmma_f32_16x16x32_bf16(
      false, a, false, b, (short)0, c, false, false);
}

// ---------------- conversion kernels ----------------
__global__ __launch_bounds__(256) void cvt_f32_bf16(
    const float* __restrict__ src, __bf16* __restrict__ dst, int n) {
  int i = blockIdx.x * blockDim.x + threadIdx.x;
  if (i < n) dst[i] = f2bf(src[i]);
}

// src [rows][cols] f32 -> dst [cols][rows] bf16   (weights -> N-major)
__global__ __launch_bounds__(256) void transpose_cvt(
    const float* __restrict__ src, __bf16* __restrict__ dst, int rows, int cols) {
  int i = blockIdx.x * blockDim.x + threadIdx.x;
  if (i < rows * cols) {
    int r = i / cols, c = i - r * cols;
    dst[(size_t)c * rows + r] = f2bf(src[i]);
  }
}

// ---------------- QKV projection GEMM ----------------
// A [MTOT x 768] bf16 row-major, Bt [2304 x 768] bf16 (N-major), bias f32.
// Writes: Q (pre-scaled by 1/sqrt(HD)) and K in [B,H,T,HD];
//         V transposed in [B,H,HD,T] so PV B-fragments load contiguously.
__global__ __launch_bounds__(128) void qkv_gemm(
    const __bf16* __restrict__ A, const __bf16* __restrict__ Bt,
    const float* __restrict__ bias,
    __bf16* __restrict__ Qb, __bf16* __restrict__ Kb, __bf16* __restrict__ Vt) {
  const int K = DD;
  int w  = threadIdx.x >> 5;                       // 4 waves: 2x2 of 64x64
  int m0 = blockIdx.x * 128 + (w >> 1) * 64;
  int n0 = blockIdx.y * 128 + (w & 1) * 64;

  v8f acc[4][4];
  for (int i = 0; i < 4; ++i) for (int j = 0; j < 4; ++j) acc[i][j] = v8f_zero();

  for (int k0 = 0; k0 < K; k0 += 32) {
    if (k0 + 32 < K) {                              // global_prefetch_b8 next slab
      __builtin_prefetch(A  + (size_t)m0 * K + k0 + 32, 0, 1);
      __builtin_prefetch(Bt + (size_t)n0 * K + k0 + 32, 0, 1);
    }
    v16bf a[4], b[4];
    for (int i = 0; i < 4; ++i) a[i] = load_a_frag (A  + (size_t)(m0 + i * 16) * K + k0, K);
    for (int j = 0; j < 4; ++j) b[j] = load_bt_frag(Bt + (size_t)(n0 + j * 16) * K + k0, K);
    for (int i = 0; i < 4; ++i)
      for (int j = 0; j < 4; ++j)
        acc[i][j] = wmma_bf16(a[i], b[j], acc[i][j]);
  }

  int lane  = threadIdx.x & 31;
  int rowhi = (lane >> 4) << 3;   // C frag: vgpr e -> row e + rowhi, col = lane%16
  int ncol  = lane & 15;
  for (int i = 0; i < 4; ++i)
    for (int j = 0; j < 4; ++j)
      for (int e = 0; e < 8; ++e) {
        int m = m0 + i * 16 + e + rowhi;
        int n = n0 + j * 16 + ncol;
        float v = acc[i][j][e] + bias[n];
        int which = n / DD, d = n - which * DD;
        int h = d >> 6, hd = d & 63;
        int bb = m / TT, t = m - bb * TT;
        size_t bhofs = ((size_t)bb * HH + h);
        if (which == 0)      Qb[(bhofs * TT + t) * HD + hd] = f2bf(v * SCALE);
        else if (which == 1) Kb[(bhofs * TT + t) * HD + hd] = f2bf(v);
        else                 Vt[(bhofs * HD + hd) * TT + t] = f2bf(v);
      }
}

// ---------------- flash attention (causal, online softmax) ----------------
// One wave owns a 32-row q tile; MASK specializes the diagonal key block.
template <bool MASK>
__device__ inline void attn_block(
    int kb, int q0, int lane, int rowhi, int ncol,
    const __bf16* __restrict__ Kp, const __bf16* __restrict__ Vtp,
    __bf16* Pw, const v16bf qf[2][2], v8f o[2][4],
    float m_[2][8], float l_[2][8]) {
  int kbase = kb * 32;

  // ---- S = Q * K^T (K^T B-frags load contiguously from row-major K) ----
  v8f s[2][2];
  for (int qi = 0; qi < 2; ++qi) { s[qi][0] = v8f_zero(); s[qi][1] = v8f_zero(); }
  for (int hc = 0; hc < 2; ++hc) {
    v16bf kt0 = load_bt_frag(Kp + (size_t)(kbase +  0) * HD + hc * 32, HD);
    v16bf kt1 = load_bt_frag(Kp + (size_t)(kbase + 16) * HD + hc * 32, HD);
    for (int qi = 0; qi < 2; ++qi) {
      s[qi][0] = wmma_bf16(qf[qi][hc], kt0, s[qi][0]);
      s[qi][1] = wmma_bf16(qf[qi][hc], kt1, s[qi][1]);
    }
  }
  if (!MASK) {                                      // prefetch next K block
    __builtin_prefetch(Kp + (size_t)(kbase + 32) * HD, 0, 1);
    __builtin_prefetch(Vtp + kbase + 32, 0, 1);
  }

  // ---- mask (diagonal block only), online softmax; row stats per-VGPR ----
  for (int qi = 0; qi < 2; ++qi) {
    float mnew[8];
    for (int e = 0; e < 8; ++e) {
      float s0 = s[qi][0][e];
      float s1 = s[qi][1][e];
      if (MASK) {
        int qrow = q0 + qi * 16 + e + rowhi;
        if (kbase + ncol      > qrow) s0 = -3.0e38f;
        if (kbase + 16 + ncol > qrow) s1 = -3.0e38f;
        s[qi][0][e] = s0; s[qi][1][e] = s1;
      }
      float mx = fmaxf(s0, s1);
      for (int d = 1; d < 16; d <<= 1) mx = fmaxf(mx, __shfl_xor(mx, d, 32));
      mnew[e] = fmaxf(m_[qi][e], mx);
    }
    for (int e = 0; e < 8; ++e) {
      float alpha = __expf(m_[qi][e] - mnew[e]);
      float p0 = __expf(s[qi][0][e] - mnew[e]);
      float p1 = __expf(s[qi][1][e] - mnew[e]);
      float rs = p0 + p1;
      for (int d = 1; d < 16; d <<= 1) rs += __shfl_xor(rs, d, 32);
      l_[qi][e] = l_[qi][e] * alpha + rs;
      m_[qi][e] = mnew[e];
      for (int nj = 0; nj < 4; ++nj) o[qi][nj][e] *= alpha;
      int prow = qi * 16 + e + rowhi;
      Pw[prow * 32 + ncol]      = f2bf(p0);
      Pw[prow * 32 + 16 + ncol] = f2bf(p1);
    }
  }

  // wave-local LDS store->load fence (no __syncthreads: divergent trip counts)
  asm volatile("s_wait_dscnt 0" ::: "memory");

  // ---- O += P * V  (V B-frags straight from global [hd][T] layout) ----
  v16bf vt[4];
  for (int nj = 0; nj < 4; ++nj)
    vt[nj] = load_bt_frag(Vtp + (size_t)(nj * 16) * TT + kbase, TT);
  for (int qi = 0; qi < 2; ++qi) {
    v16bf pf = load_a_frag(Pw + (qi * 16) * 32, 32);
    for (int nj = 0; nj < 4; ++nj)
      o[qi][nj] = wmma_bf16(pf, vt[nj], o[qi][nj]);
  }
}

__global__ __launch_bounds__(256) void flash_attn(
    const __bf16* __restrict__ Qb, const __bf16* __restrict__ Kb,
    const __bf16* __restrict__ Vt, __bf16* __restrict__ Ob /* [B*T, D] bf16 */) {
  __shared__ __bf16 ldsP[8][32 * 32];    // per-wave P tile (32 q x 32 k)

  int w    = threadIdx.x >> 5;
  int lane = threadIdx.x & 31;
  int bh   = blockIdx.y;                 // 0..23
  int b    = bh / HH, h = bh - b * HH;
  int qt   = blockIdx.x * 8 + w;         // q tile id, 0..127
  int q0   = qt * 32;

  const __bf16* Qp  = Qb + (size_t)bh * TT * HD;
  const __bf16* Kp  = Kb + (size_t)bh * TT * HD;
  const __bf16* Vtp = Vt + (size_t)bh * HD * TT;
  __bf16* Pw = ldsP[w];

  // Q fragments (pre-scaled by 1/sqrt(HD) at projection time)
  v16bf qf[2][2];
  for (int qi = 0; qi < 2; ++qi)
    for (int hc = 0; hc < 2; ++hc)
      qf[qi][hc] = load_a_frag(Qp + (size_t)(q0 + qi * 16) * HD + hc * 32, HD);

  v8f o[2][4];
  float m_[2][8], l_[2][8];
  for (int qi = 0; qi < 2; ++qi) {
    for (int nj = 0; nj < 4; ++nj) o[qi][nj] = v8f_zero();
    for (int e = 0; e < 8; ++e) { m_[qi][e] = -3.0e38f; l_[qi][e] = 0.0f; }
  }

  int rowhi = (lane >> 4) << 3;
  int ncol  = lane & 15;

  for (int kb = 0; kb < qt; ++kb)        // interior blocks: no masking
    attn_block<false>(kb, q0, lane, rowhi, ncol, Kp, Vtp, Pw, qf, o, m_, l_);
  attn_block<true>(qt, q0, lane, rowhi, ncol, Kp, Vtp, Pw, qf, o, m_, l_);

  // ---- normalize and write attention output as bf16 [B*T, D] ----
  for (int qi = 0; qi < 2; ++qi)
    for (int e = 0; e < 8; ++e) {
      float inv = 1.0f / l_[qi][e];
      int t = q0 + qi * 16 + e + rowhi;
      size_t rowoff = ((size_t)b * TT + t) * DD + (size_t)h * HD;
      for (int nj = 0; nj < 4; ++nj)
        Ob[rowoff + nj * 16 + ncol] = f2bf(o[qi][nj][e] * inv);
    }
}

// ---------------- output projection GEMM (f32 out) ----------------
__global__ __launch_bounds__(128) void out_gemm(
    const __bf16* __restrict__ A, const __bf16* __restrict__ Bt,
    const float* __restrict__ bias, float* __restrict__ out) {
  const int K = DD;
  int w  = threadIdx.x >> 5;
  int m0 = blockIdx.x * 128 + (w >> 1) * 64;
  int n0 = blockIdx.y * 128 + (w & 1) * 64;

  v8f acc[4][4];
  for (int i = 0; i < 4; ++i) for (int j = 0; j < 4; ++j) acc[i][j] = v8f_zero();

  for (int k0 = 0; k0 < K; k0 += 32) {
    if (k0 + 32 < K) {
      __builtin_prefetch(A  + (size_t)m0 * K + k0 + 32, 0, 1);
      __builtin_prefetch(Bt + (size_t)n0 * K + k0 + 32, 0, 1);
    }
    v16bf a[4], bfr[4];
    for (int i = 0; i < 4; ++i) a[i]   = load_a_frag (A  + (size_t)(m0 + i * 16) * K + k0, K);
    for (int j = 0; j < 4; ++j) bfr[j] = load_bt_frag(Bt + (size_t)(n0 + j * 16) * K + k0, K);
    for (int i = 0; i < 4; ++i)
      for (int j = 0; j < 4; ++j)
        acc[i][j] = wmma_bf16(a[i], bfr[j], acc[i][j]);
  }

  int lane  = threadIdx.x & 31;
  int rowhi = (lane >> 4) << 3;
  int ncol  = lane & 15;
  for (int i = 0; i < 4; ++i)
    for (int j = 0; j < 4; ++j)
      for (int e = 0; e < 8; ++e) {
        int m = m0 + i * 16 + e + rowhi;
        int n = n0 + j * 16 + ncol;
        out[(size_t)m * DD + n] = acc[i][j][e] + bias[n];
      }
}

// ---------------- launch ----------------
extern "C" void kernel_launch(void* const* d_in, const int* in_sizes, int n_in,
                              void* d_out, int out_size, void* d_ws, size_t ws_size,
                              hipStream_t stream) {
  const float* x    = (const float*)d_in[0];
  const float* Wqkv = (const float*)d_in[1];
  const float* bqkv = (const float*)d_in[2];
  const float* Wout = (const float*)d_in[3];
  const float* bout = (const float*)d_in[4];
  float* out = (float*)d_out;

  // workspace layout (bf16): x | Wqkv^T | Wout^T | Q | K | V^T | attn_out (~68 MB)
  __bf16* ws = (__bf16*)d_ws;
  size_t off = 0;
  __bf16* xb     = ws + off; off += (size_t)MTOT * DD;
  __bf16* WqkvT  = ws + off; off += (size_t)NQKV * DD;
  __bf16* WoutT  = ws + off; off += (size_t)DD * DD;
  __bf16* Qb     = ws + off; off += (size_t)MTOT * DD;
  __bf16* Kb     = ws + off; off += (size_t)MTOT * DD;
  __bf16* Vt     = ws + off; off += (size_t)MTOT * DD;
  __bf16* attn   = ws + off; off += (size_t)MTOT * DD;

  int nx = MTOT * DD;
  cvt_f32_bf16<<<(nx + 255) / 256, 256, 0, stream>>>(x, xb, nx);
  transpose_cvt<<<(DD * NQKV + 255) / 256, 256, 0, stream>>>(Wqkv, WqkvT, DD, NQKV);
  transpose_cvt<<<(DD * DD + 255) / 256, 256, 0, stream>>>(Wout, WoutT, DD, DD);

  qkv_gemm<<<dim3(MTOT / 128, NQKV / 128), 128, 0, stream>>>(xb, WqkvT, bqkv, Qb, Kb, Vt);

  flash_attn<<<dim3(TT / 256, BB * HH), 256, 0, stream>>>(Qb, Kb, Vt, attn);

  out_gemm<<<dim3(MTOT / 128, DD / 128), 128, 0, stream>>>(attn, WoutT, bout, out);
}
